// LSTM_4904852652924
// MI455X (gfx1250) — matrix operand order repaired and checked by reading, not compiled, not run on hardware
//
#include <hip/hip_runtime.h>

// ---------------------------------------------------------------------------
// LSTM on MI455X (gfx1250, wave32).
// Per step: gates[B,4H] = [h|x_t] @ W^T + b  via v_wmma_f32_16x16x32_bf16.
// The 8 waves of a block share one hidden n-tile; the 4 gate weight fragments
// (4KB / K-step) are staged into LDS with one GLOBAL_LOAD_ASYNC_TO_LDS_B128
// per block-iteration (256 lanes x 16B), double-buffered. Loop is structured
// barrier-at-top so the next stage is issued immediately after the barrier and
// its latency is covered by the 4 WMMAs + ds/global loads of the current step.
// Raw split barriers (s_barrier_signal/-wait) are used so the register-
// pipelined A-fragment prefetch can stay in flight across the barrier.
// ---------------------------------------------------------------------------

typedef __attribute__((ext_vector_type(16))) __bf16 v16bf;
typedef __attribute__((ext_vector_type(8)))  float  v8f;

#define B_ 256
#define S_ 512
#define D_ 256
#define H_ 512
#define K_ (H_ + D_)      // 768
#define KSTEPS (K_ / 32)  // 24

// ---------------- helpers ----------------

__device__ __forceinline__ v8f wmma_bf16(v16bf a, v16bf b, v8f c) {
  // (neg_a, A, neg_b, B, c_mod, C, reuse_a, reuse_b)
  return __builtin_amdgcn_wmma_f32_16x16x32_bf16(false, a, false, b,
                                                 (short)0, c, false, false);
}

__device__ __forceinline__ float sigm(float v) {
  return 1.0f / (1.0f + __expf(-v));
}
__device__ __forceinline__ float tanh_fast(float v) {
  float e = __expf(2.0f * v);
  return 1.0f - 2.0f / (e + 1.0f);  // -> -1 / +1 correctly at extremes
}

// Pack 16 f32 (two 8-float chunks) into the bf16 A-fragment register layout.
__device__ __forceinline__ v16bf cvt_a(float4 q0, float4 q1, float4 q2, float4 q3) {
  v16bf a;
  a[0]=(__bf16)q0.x;  a[1]=(__bf16)q0.y;  a[2]=(__bf16)q0.z;  a[3]=(__bf16)q0.w;
  a[4]=(__bf16)q1.x;  a[5]=(__bf16)q1.y;  a[6]=(__bf16)q1.z;  a[7]=(__bf16)q1.w;
  a[8]=(__bf16)q2.x;  a[9]=(__bf16)q2.y;  a[10]=(__bf16)q2.z; a[11]=(__bf16)q2.w;
  a[12]=(__bf16)q3.x; a[13]=(__bf16)q3.y; a[14]=(__bf16)q3.z; a[15]=(__bf16)q3.w;
  return a;
}

__device__ __forceinline__ void wait_async0() {
  asm volatile("s_wait_asynccnt 0x0" ::: "memory");
}

// Raw workgroup split-barrier: no compiler-inserted loadcnt drain, so the
// register-pipelined global A prefetch stays in flight across it. Cross-wave
// LDS ordering is guaranteed by s_wait_asynccnt (writes) and the dscnt waits
// the consuming WMMAs already force (reads).
__device__ __forceinline__ void wg_barrier() {
  asm volatile("s_barrier_signal -1\n\t"
               "s_barrier_wait -1" ::: "memory");
}

// ---------------- prologue kernels ----------------

__global__ __launch_bounds__(256) void cvt_weights(
    const float* __restrict__ Wf, const float* __restrict__ Wi,
    const float* __restrict__ Wo, const float* __restrict__ Wg,
    unsigned short* __restrict__ out) {
  int i = blockIdx.x * blockDim.x + threadIdx.x;
  const int n = H_ * K_;
  if (i >= n) return;
  out[i]         = __builtin_bit_cast(unsigned short, (__bf16)Wf[i]);
  out[i + n]     = __builtin_bit_cast(unsigned short, (__bf16)Wi[i]);
  out[i + 2 * n] = __builtin_bit_cast(unsigned short, (__bf16)Wo[i]);
  out[i + 3 * n] = __builtin_bit_cast(unsigned short, (__bf16)Wg[i]);
}

__global__ __launch_bounds__(256) void zero_f32(float* __restrict__ p, int n) {
  int i = blockIdx.x * blockDim.x + threadIdx.x;
  if (i < n) p[i] = 0.0f;
}

// ---------------- per-timestep kernel ----------------
// 64 blocks x 8 waves. Block -> one hidden n-tile (n0) and 8 batch m-tiles.

__global__ __launch_bounds__(256) void lstm_step(
    const float* __restrict__ x,          // [B, S, D]
    const unsigned short* __restrict__ W, // bf16 [4][H][K], gate order f,i,o,g
    const float* __restrict__ bf, const float* __restrict__ bi,
    const float* __restrict__ bo, const float* __restrict__ bg,
    float* __restrict__ h,                // [B, H] state
    float* __restrict__ c,                // [B, H] state
    float* __restrict__ out,              // [S, B, H]
    int t) {
  // Double-buffered weight stage: 2 x 4 gates x (16 rows x 32 K) bf16 = 8KB.
  __shared__ __attribute__((aligned(32))) unsigned short smem[2][4][16 * 32];

  const int tid  = threadIdx.x;
  const int lane = tid & 31;
  const int wave = tid >> 5;
  const int n0   = (blockIdx.x >> 1) << 4;                  // hidden tile base
  const int m0   = (((blockIdx.x & 1) << 3) + wave) << 4;   // batch tile base

  const int  ln     = lane & 15;
  const bool hiHalf = lane >= 16;
  const int  m      = m0 + ln;            // A row (batch) for this lane
  const int  n      = n0 + ln;            // B column (hidden) for this lane
  const int  kbA    = hiHalf ? 16 : 0;    // A half-wave K base
  const int  kbB    = hiHalf ? 16 : 0;    // B half-wave K base

  // Async stage: thread tid copies one 16B chunk of one gate row.
  const int sg    = tid >> 6;             // gate 0..3
  const int sj    = tid & 63;
  const int srow  = sj >> 2;              // 0..15
  const int schk  = (sj & 3) << 3;        // ushort offset of 16B chunk
  const unsigned short* sbase =
      W + (size_t)sg * H_ * K_ + (size_t)(n0 + srow) * K_ + schk;
  auto stage = [&](int kt, int buf) {
    const unsigned short* gsrc = sbase + kt * 32;
    unsigned laddr = (unsigned)(uintptr_t)&smem[buf][sg][srow * 32 + schk];
    unsigned long long gaddr = (unsigned long long)(uintptr_t)gsrc;
    asm volatile("global_load_async_to_lds_b128 %0, %1, off"
                 :: "v"(laddr), "v"(gaddr) : "memory");
  };

  // Bias depends only on n (same for all 8 acc rows).
  const float bFv = bf[n], bIv = bi[n], bOv = bo[n], bGv = bg[n];
  v8f accF, accI, accO, accG;
#pragma unroll
  for (int r = 0; r < 8; ++r) { accF[r]=bFv; accI[r]=bIv; accO[r]=bOv; accG[r]=bGv; }

  const float* hrow = h + m * H_;
  const float* xrow = x + ((size_t)m * S_ + t) * D_;
  auto aptr = [&](int kt) -> const float* {
    return (kt < 16) ? (hrow + kt * 32 + kbA) : (xrow + (kt - 16) * 32 + kbA);
  };

  // Prologue: stage K-step 0, prefetch A(0), drain own async copy.
  float4 a0, a1, a2, a3;
  {
    const float4* f = (const float4*)aptr(0);
    a0 = f[0]; a1 = f[1]; a2 = f[2]; a3 = f[3];
  }
  stage(0, 0);
  wait_async0();

#pragma unroll
  for (int kt = 0; kt < KSTEPS; ++kt) {
    const int buf = kt & 1;
    wg_barrier();                 // stage kt visible to all waves
    if (kt < KSTEPS - 1)
      stage(kt + 1, buf ^ 1);     // issue next copy first: latency covered below
    // B fragments from LDS (ds_load_b128 x2 per gate).
    v16bf fF = *(const v16bf*)&smem[buf][0][ln * 32 + kbB];
    v16bf fI = *(const v16bf*)&smem[buf][1][ln * 32 + kbB];
    v16bf fO = *(const v16bf*)&smem[buf][2][ln * 32 + kbB];
    v16bf fG = *(const v16bf*)&smem[buf][3][ln * 32 + kbB];
    v16bf a = cvt_a(a0, a1, a2, a3);
    if (kt < KSTEPS - 1) {
      // Prefetch next A fragment (global f32); stays in flight past barrier.
      const float4* f = (const float4*)aptr(kt + 1);
      a0 = f[0]; a1 = f[1]; a2 = f[2]; a3 = f[3];
    }
    accF = wmma_bf16(a, fF, accF);
    accI = wmma_bf16(a, fI, accI);
    accO = wmma_bf16(a, fO, accO);
    accG = wmma_bf16(a, fG, accG);
    if (kt < KSTEPS - 1)
      wait_async0();              // own copy of next stage landed
  }

  // ---- fused LSTM cell update.
  // D layout: VGPR r, lane L -> element [M = r + 8*(L>=16)][N = L&15].
  float* outT = out + (size_t)t * (B_ * H_);
#pragma unroll
  for (int r = 0; r < 8; ++r) {
    const float fv = sigm(accF[r]);
    const float iv = sigm(accI[r]);
    const float ov = sigm(accO[r]);
    const float gv = tanh_fast(accG[r]);
    const int b   = m0 + r + (hiHalf ? 8 : 0);
    const int idx = b * H_ + n0 + ln;
    const float cn = fv * c[idx] + iv * gv;
    c[idx] = cn;
    const float hn = ov * tanh_fast(cn);
    h[idx] = hn;
    outT[idx] = hn;
  }
}

// ---------------- launch ----------------

extern "C" void kernel_launch(void* const* d_in, const int* in_sizes, int n_in,
                              void* d_out, int out_size, void* d_ws, size_t ws_size,
                              hipStream_t stream) {
  (void)in_sizes; (void)n_in; (void)out_size; (void)ws_size;
  const float* x  = (const float*)d_in[0];
  const float* Wf = (const float*)d_in[1];
  const float* bf = (const float*)d_in[2];
  const float* Wi = (const float*)d_in[3];
  const float* bi = (const float*)d_in[4];
  const float* Wo = (const float*)d_in[5];
  const float* bo = (const float*)d_in[6];
  const float* Wg = (const float*)d_in[7];
  const float* bg = (const float*)d_in[8];
  float* out = (float*)d_out;

  // Workspace layout: [bf16 weights 4*H*K | h (B*H f32) | c (B*H f32)] ~4.2 MB
  unsigned short* Wbf = (unsigned short*)d_ws;
  float* h = (float*)((char*)d_ws + (size_t)4 * H_ * K_ * sizeof(unsigned short));
  float* c = h + B_ * H_;

  const int nW = H_ * K_;
  cvt_weights<<<(nW + 255) / 256, 256, 0, stream>>>(Wf, Wi, Wo, Wg, Wbf);
  zero_f32<<<(2 * B_ * H_ + 255) / 256, 256, 0, stream>>>(h, 2 * B_ * H_);

  for (int t = 0; t < S_; ++t) {
    lstm_step<<<64, 256, 0, stream>>>(x, Wbf, bf, bi, bo, bg, h, c, out, t);
  }
}
